// GatedAttentionPooling_12266426597825
// MI455X (gfx1250) — matrix operand order repaired and checked by reading, not compiled, not run on hardware
//
#include <hip/hip_runtime.h>
#include <hip/hip_bf16.h>
#include <math.h>

// ---------------------------------------------------------------------------
// Gated attention MIL pooling for MI455X (gfx1250, wave32).
//   Phase 1: per-row attention scores via bf16 WMMA (V_WMMA_F32_16X16X32_BF16)
//   Phase 2: ragged segment softmax + weighted pooling (bandwidth-bound)
// ---------------------------------------------------------------------------

typedef __attribute__((ext_vector_type(16))) __bf16 v16bf;
typedef __attribute__((ext_vector_type(8)))  float  v8f;

#define FDIM 128   // feature dim (= K of the GEMM)
#define HID  256   // hidden dim of V/U

union BFu { v16bf v; unsigned int u[8]; };

__device__ __forceinline__ unsigned int pack_bf2(float a, float b) {
    // round-to-nearest-even f32 -> bf16, packed pair
    unsigned int ua = __float_as_uint(a);
    unsigned int ub = __float_as_uint(b);
    ua = (ua + 0x7FFFu + ((ua >> 16) & 1u)) >> 16;
    ub = (ub + 0x7FFFu + ((ub >> 16) & 1u)) >> 16;
    return (ub << 16) | (ua & 0xFFFFu);
}

// Branch-free transcendentals; tanh uses the CDNA5 v_tanh_f32 when available
// (TRANS32 op -> co-executes with XDL WMMA), else exp/rcp formulation.
__device__ __forceinline__ float fast_tanh(float x) {
#if __has_builtin(__builtin_amdgcn_tanhf)
    return __builtin_amdgcn_tanhf(x);
#else
    const float e = __expf(2.0f * x);              // v_exp_f32
    return 1.0f - 2.0f * __builtin_amdgcn_rcpf(e + 1.0f);  // v_rcp_f32
#endif
}

__device__ __forceinline__ float fast_sigmoid(float x) {
    return __builtin_amdgcn_rcpf(1.0f + __expf(-x));
}

// ---------------------------------------------------------------------------
// Kernel 1: A[n] = ww . ( tanh(Vw x_n + Vb) * sigmoid(Uw x_n + Ub) ) + wb
// 256 threads = 8 waves. Wave w owns hidden column tiles {2w, 2w+1} for BOTH
// V and U (4 accumulator tiles). B fragments live in registers for the whole
// kernel; x tiles are staged through LDS as bf16 and gathered in the contiguous
// k order (k = 16*half + s) also used for B.
// ---------------------------------------------------------------------------
__global__ __launch_bounds__(256) void gated_attn_scores(
    const float* __restrict__ x,
    const float* __restrict__ Vw, const float* __restrict__ Vb,
    const float* __restrict__ Uw, const float* __restrict__ Ub,
    const float* __restrict__ ww, const float* __restrict__ wb,
    float* __restrict__ A, int N, int ntiles)
{
    __shared__ unsigned int xlds[16][68];   // 16 rows x 64 bf16-pairs (+pad)
    __shared__ float a_part[8][16];         // per-wave partial scores

    const int tid  = threadIdx.x;
    const int lane = tid & 31;
    const int wave = tid >> 5;
    const int m    = lane & 15;   // row within tile / hidden-col within tile
    const int half = lane >> 4;   // lane half selects K sub-block

    const int n0 = (wave*2 + 0)*16 + m;     // hidden index, column tile 0
    const int n1 = (wave*2 + 1)*16 + m;     // hidden index, column tile 1

    // ---- Persistent B fragments: B = Wᵀ (K=128 x N=256), bf16.
    // 32x16 B, contiguous k order (same as A): VGPR v, lane L -> {B[2v+16*half][n], B[2v+1+16*half][n]}
    // B[k][n] = W[n][k]  (W row-major [HID][FDIM])
    v16bf Bf[2][2][4];                      // [V|U][coltile][kchunk]
    const float* Wmat[2] = { Vw, Uw };
    #pragma unroll
    for (int mat = 0; mat < 2; ++mat) {
        #pragma unroll
        for (int ct = 0; ct < 2; ++ct) {
            const float* wrow = Wmat[mat] + (long)(ct ? n1 : n0) * FDIM;
            #pragma unroll
            for (int kc = 0; kc < 4; ++kc) {
                BFu u;
                #pragma unroll
                for (int v = 0; v < 8; ++v) {
                    const int k = kc*32 + 2*v + 16*half;
                    const float2 p = *(const float2*)(wrow + k);
                    u.u[v] = pack_bf2(p.x, p.y);
                }
                Bf[mat][ct][kc] = u.v;
            }
        }
    }
    const float vb0 = Vb[n0], vb1 = Vb[n1];
    const float ub0 = Ub[n0], ub1 = Ub[n1];
    const float w0  = ww[n0], w1  = ww[n1];
    const float wbias = wb[0];

    for (long tile = blockIdx.x; tile < ntiles; tile += gridDim.x) {
        const long row0 = tile * 16;
        __syncthreads();  // previous iteration fully consumed xlds / a_part

        // ---- Stage 16 x 128 f32 -> bf16 pairs into LDS.
        // float4 (b128) global loads, uint2 (b64) LDS stores: 2 iters/thread.
        #pragma unroll
        for (int p = tid; p < 16*32; p += 256) {
            const int r = p >> 5, c4 = p & 31;    // c4: which float4 of the row
            const long row = row0 + r;
            float4 xv = make_float4(0.0f, 0.0f, 0.0f, 0.0f);
            if (row < N) xv = *(const float4*)(x + row*FDIM + c4*4);
            uint2 pk;
            pk.x = pack_bf2(xv.x, xv.y);
            pk.y = pack_bf2(xv.z, xv.w);
            *(uint2*)&xlds[r][c4*2] = pk;
        }
        // Prefetch next tile this block will touch (global_prefetch_b8)
        {
            const long nrow0 = row0 + (long)gridDim.x * 16;
            if (tid < 64 && nrow0 + (tid >> 2) < N)
                __builtin_prefetch(x + (nrow0 + (tid >> 2))*FDIM + (long)(tid & 3)*32, 0, 0);
        }
        __syncthreads();

        // ---- Gather A fragments, contiguous k order (same as B):
        // VGPR v, half h -> K pair starting at k = 2v + 16h
        v16bf Af[4];
        #pragma unroll
        for (int kc = 0; kc < 4; ++kc) {
            BFu u;
            #pragma unroll
            for (int v = 0; v < 8; ++v) {
                const int kh = kc*16 + v + half*8; // pair index
                u.u[v] = xlds[m][kh];
            }
            Af[kc] = u.v;
        }

        // ---- 16 WMMAs per wave: 4 output tiles x 4 K-chunks (K = 128)
        v8f accV0 = {}, accV1 = {}, accU0 = {}, accU1 = {};
        #pragma unroll
        for (int kc = 0; kc < 4; ++kc) {
            accV0 = __builtin_amdgcn_wmma_f32_16x16x32_bf16(false, Af[kc], false, Bf[0][0][kc], (short)0, accV0, false, false);
            accV1 = __builtin_amdgcn_wmma_f32_16x16x32_bf16(false, Af[kc], false, Bf[0][1][kc], (short)0, accV1, false, false);
            accU0 = __builtin_amdgcn_wmma_f32_16x16x32_bf16(false, Af[kc], false, Bf[1][0][kc], (short)0, accU0, false, false);
            accU1 = __builtin_amdgcn_wmma_f32_16x16x32_bf16(false, Af[kc], false, Bf[1][1][kc], (short)0, accU1, false, false);
        }

        // ---- Gate + dot(ww). C/D layout: VGPR r, lanes 0-15 -> row r (cols=lane),
        // lanes 16-31 -> row r+8. Reduce over 16 columns within each lane half.
        float part[8];
        #pragma unroll
        for (int r = 0; r < 8; ++r) {
            const float g0 = fast_tanh(accV0[r] + vb0) * fast_sigmoid(accU0[r] + ub0);
            const float g1 = fast_tanh(accV1[r] + vb1) * fast_sigmoid(accU1[r] + ub1);
            part[r] = g0*w0 + g1*w1;
        }
        #pragma unroll
        for (int off = 8; off >= 1; off >>= 1) {
            #pragma unroll
            for (int r = 0; r < 8; ++r)
                part[r] += __shfl_xor(part[r], off, 32);
        }
        if (m == 0) {  // lane 0 (rows 0-7) and lane 16 (rows 8-15)
            #pragma unroll
            for (int r = 0; r < 8; ++r)
                a_part[wave][r + 8*half] = part[r];
        }
        __syncthreads();

        // ---- Deterministic cross-wave combine + store
        if (tid < 16 && row0 + tid < N) {
            float s = wbias;
            #pragma unroll
            for (int w = 0; w < 8; ++w) s += a_part[w][tid];
            A[row0 + tid] = s;
        }
    }
}

// ---------------------------------------------------------------------------
// Kernel 2: per-bag numerically-stable softmax over A + weighted pooling of x.
// One 128-thread block per bag; thread t owns feature t (coalesced x reads).
// ---------------------------------------------------------------------------
__global__ __launch_bounds__(128) void segment_softmax_pool(
    const float* __restrict__ x, const int* __restrict__ seg,
    const float* __restrict__ A, float* __restrict__ out, int N)
{
    const int b = blockIdx.x;
    const int t = threadIdx.x;
    __shared__ float red[128];
    __shared__ float wl[128];
    __shared__ int bounds[2];

    if (t < 2) {  // lower_bound(seg, b) and lower_bound(seg, b+1) on sorted seg
        const int target = b + t;
        int lo = 0, hi = N;
        while (lo < hi) { const int mid = (lo + hi) >> 1;
                          if (seg[mid] < target) lo = mid + 1; else hi = mid; }
        bounds[t] = lo;
    }
    __syncthreads();
    const int lo = bounds[0], hi = bounds[1];

    // segment max
    float mx = -INFINITY;
    for (int i = lo + t; i < hi; i += 128) mx = fmaxf(mx, A[i]);
    red[t] = mx; __syncthreads();
    for (int s = 64; s; s >>= 1) { if (t < s) red[t] = fmaxf(red[t], red[t + s]); __syncthreads(); }
    mx = red[0]; __syncthreads();

    // segment exp-sum
    float sm = 0.0f;
    for (int i = lo + t; i < hi; i += 128) sm += __expf(A[i] - mx);
    red[t] = sm; __syncthreads();
    for (int s = 64; s; s >>= 1) { if (t < s) red[t] += red[t + s]; __syncthreads(); }
    const float inv = 1.0f / (red[0] + 1e-9f);
    __syncthreads();

    // weighted pooling, 128-row chunks with weights staged in LDS
    float acc = 0.0f;
    for (int cs = lo; cs < hi; cs += 128) {
        const int i = cs + t;
        if (i < hi) wl[t] = __expf(A[i] - mx) * inv;
        __syncthreads();
        const int cnt = min(128, hi - cs);
        const float* xp = x + (long)cs * FDIM + t;
        for (int j = 0; j < cnt; ++j)
            acc = fmaf(xp[(long)j * FDIM], wl[j], acc);
        __syncthreads();
    }
    out[(long)b * FDIM + t] = acc;  // empty bags -> 0, matching segment_sum
}

// ---------------------------------------------------------------------------
extern "C" void kernel_launch(void* const* d_in, const int* in_sizes, int n_in,
                              void* d_out, int out_size, void* d_ws, size_t ws_size,
                              hipStream_t stream)
{
    const float* x   = (const float*)d_in[0];
    const int*   seg = (const int*)  d_in[1];
    // d_in[2] = batch_size (scalar on device; B derived from out_size instead)
    const float* Vw  = (const float*)d_in[3];
    const float* Vb  = (const float*)d_in[4];
    const float* Uw  = (const float*)d_in[5];
    const float* Ub  = (const float*)d_in[6];
    const float* ww  = (const float*)d_in[7];
    const float* wb  = (const float*)d_in[8];
    float* out = (float*)d_out;
    float* A   = (float*)d_ws;          // N floats of scratch (2 MB)

    const int N = in_sizes[0] / FDIM;   // 500000
    const int B = out_size / FDIM;      // 4096
    const int ntiles = (N + 15) / 16;

    const int grid1 = ntiles < 2048 ? ntiles : 2048;
    gated_attn_scores<<<grid1, 256, 0, stream>>>(x, Vw, Vb, Uw, Ub, ww, wb, A, N, ntiles);
    segment_softmax_pool<<<B, 128, 0, stream>>>(x, seg, A, out, N);
}
